// AttentionHead_36962488549940
// MI455X (gfx1250) — matrix operand order repaired and checked by reading, not compiled, not run on hardware
//
#include <hip/hip_runtime.h>

typedef __attribute__((ext_vector_type(16))) _Float16 v16h;
typedef __attribute__((ext_vector_type(8)))  _Float16 v8h;
typedef __attribute__((ext_vector_type(8)))  float    v8f;
typedef __attribute__((ext_vector_type(4)))  float    v4f;
typedef __attribute__((ext_vector_type(4)))  unsigned u32x4;
typedef __attribute__((ext_vector_type(8)))  int      i32x8;
typedef __attribute__((ext_vector_type(4)))  int      i32x4;

#define B_DIM 32
#define T_DIM 768
#define C_DIM 256
#define H_DIM 64
#define QTILES (T_DIM / 16)      // 48 query tiles per batch
#define NEG_INF (-1e30f)

#if defined(__HIP_DEVICE_COMPILE__) && __has_builtin(__builtin_amdgcn_tensor_load_to_lds)
#define HAVE_TDM 1
#endif

__device__ __forceinline__ v8f wmma16(v16h a, v16h b, v8f c) {
  // D = A(16x32 f16) * B(32x16 f16) + C(16x16 f32)
  return __builtin_amdgcn_wmma_f32_16x16x32_f16(
      false, a, false, b, (short)0, c, false, false);
}

// ---------------------------------------------------------------------------
// Stage 0: one-shot weight prep. wT[w][n][kk] = (f16) w[kk*64 + n]
// (transposed so projection B operands are contiguous 32B runs along kk)
// ---------------------------------------------------------------------------
__global__ __launch_bounds__(256) void prep_w(const float* __restrict__ wq,
                                              const float* __restrict__ wk,
                                              const float* __restrict__ wv,
                                              _Float16* __restrict__ wt) {
  int idx = blockIdx.x * 256 + threadIdx.x;          // over 3*64*256
  if (idx >= 3 * H_DIM * C_DIM) return;
  int w   = idx / (H_DIM * C_DIM);
  int rem = idx % (H_DIM * C_DIM);
  int n   = rem / C_DIM;
  int kk  = rem % C_DIM;
  const float* src = (w == 0) ? wq : (w == 1) ? wk : wv;
  wt[idx] = (_Float16)src[kk * H_DIM + n];
}

// ---------------------------------------------------------------------------
// Stage 1: Q/K/V projection. One wave per 16 rows of x.
// A: x rows as float4 loads + cvt.  B: contiguous v16h from transposed wT.
// Q,K stored [token][h]; V stored transposed [b][h][t] for the PV GEMM.
// ---------------------------------------------------------------------------
__global__ __launch_bounds__(32) void qkv_proj(const float* __restrict__ x,
                                               const _Float16* __restrict__ wt,
                                               _Float16* __restrict__ qo,
                                               _Float16* __restrict__ ko,
                                               _Float16* __restrict__ vt) {
  const int lane  = threadIdx.x;
  const int tile  = blockIdx.x;
  const int col   = lane & 15;
  const int hi    = lane >> 4;
  const int kbase = hi * 8;            // unused: A is loaded at khalf too
  const int khalf = hi * 16;           // A/B-operand K base (contiguous k = 16h + s)

  v8f zero = {};
  v8f acc[3][4];
  for (int w = 0; w < 3; ++w)
    for (int ct = 0; ct < 4; ++ct) acc[w][ct] = zero;

  const float* xr = x + (size_t)(tile * 16 + col) * C_DIM;
  for (int c = 0; c < 8; ++c) {        // K = 256 -> 8 chunks of 32
    const float* xc = xr + c * 32;
    v4f x0 = *(const v4f*)(xc + khalf);
    v4f x1 = *(const v4f*)(xc + khalf + 4);
    v4f x2 = *(const v4f*)(xc + khalf + 8);
    v4f x3 = *(const v4f*)(xc + khalf + 12);
    v16h a;
    #pragma unroll
    for (int j = 0; j < 4; ++j) {
      a[j]      = (_Float16)x0[j];
      a[4 + j]  = (_Float16)x1[j];
      a[8 + j]  = (_Float16)x2[j];
      a[12 + j] = (_Float16)x3[j];
    }
    #pragma unroll
    for (int w = 0; w < 3; ++w)
      #pragma unroll
      for (int ct = 0; ct < 4; ++ct) {
        v16h bm = *(const v16h*)(wt + (size_t)(w * H_DIM + ct * 16 + col) * C_DIM
                                    + c * 32 + khalf);
        acc[w][ct] = wmma16(a, bm, acc[w][ct]);
      }
  }

  const int bb = (tile * 16) / T_DIM;          // batch (tiles never straddle)
  const int tb = (tile * 16) % T_DIM;
  #pragma unroll
  for (int ct = 0; ct < 4; ++ct)
    #pragma unroll
    for (int r = 0; r < 8; ++r) {
      int g = tile * 16 + r + 8 * hi;          // global token row
      int h = ct * 16 + col;                   // head dim
      qo[(size_t)g * H_DIM + h] = (_Float16)acc[0][ct][r];
      ko[(size_t)g * H_DIM + h] = (_Float16)acc[1][ct][r];
      vt[(size_t)(bb * H_DIM + h) * T_DIM + tb + r + 8 * hi] = (_Float16)acc[2][ct][r];
    }
}

// ---------------------------------------------------------------------------
// Stage 2: flash attention. One wave per (batch, 16-query tile).
// 32 keys / iteration: 4 score WMMAs + 4 PV WMMAs (full 16x32 P operand).
// K pair-tile staged to LDS via TDM tensor_load_to_lds when available.
// ---------------------------------------------------------------------------
__global__ __launch_bounds__(32) void attn(const _Float16* __restrict__ qm,
                                           const _Float16* __restrict__ km,
                                           const _Float16* __restrict__ vt,
                                           float* __restrict__ out) {
  __shared__ _Float16 pbuf[16 * 32];       // P tile (queries x 32 keys)
  __shared__ _Float16 kbuf[32 * 64];       // K pair-tile staged by TDM

  const int lane  = threadIdx.x;
  const int b     = blockIdx.x / QTILES;
  const int qt    = blockIdx.x % QTILES;
  const int col   = lane & 15;
  const int hi    = lane >> 4;
  const int kbase = hi * 8;
  const int khalf = hi * 16;

  // Q tile as A operand (two 32-wide head-dim chunks)
  const _Float16* Q = qm + (size_t)(b * T_DIM + qt * 16) * H_DIM;
  v16h qa[2];
  #pragma unroll
  for (int c = 0; c < 2; ++c) {
    const _Float16* qr = Q + col * H_DIM + c * 32;
    v8h lo  = *(const v8h*)(qr + khalf);
    v8h hi8 = *(const v8h*)(qr + khalf + 8);
    #pragma unroll
    for (int j = 0; j < 8; ++j) { qa[c][j] = lo[j]; qa[c][8 + j] = hi8[j]; }
  }

  v8f zero = {};
  v8f o[4];
  float m_run[8], l_run[8];
  #pragma unroll
  for (int r = 0; r < 8; ++r) { m_run[r] = NEG_INF; l_run[r] = 0.0f; }
  #pragma unroll
  for (int ct = 0; ct < 4; ++ct) o[ct] = zero;

  const float scale = 0.125f;                  // 1/sqrt(64)
  const int npairs = (qt + 2) >> 1;            // ceil((qt+1)/2) 32-key steps

#ifdef HAVE_TDM
  const unsigned lds_off = (unsigned)(unsigned long long)(void*)kbuf; // flat->LDS: low 32 bits
#endif

  for (int p = 0; p < npairs; ++p) {
    const int k0 = p * 32;

#ifdef HAVE_TDM
    {
      // Tensor DMA: 32x64 f16 tile, row stride 64 elements, into kbuf.
      unsigned long long ga =
          (unsigned long long)(km + (size_t)(b * T_DIM + k0) * H_DIM);
      u32x4 g0 = {};
      g0[0] = 1u;                                        // count = 1
      g0[1] = lds_off;                                   // lds_addr
      g0[2] = (unsigned)(ga & 0xFFFFFFFFu);              // global_addr[31:0]
      g0[3] = (unsigned)((ga >> 32) & 0x1FFFFFFu) | (2u << 30);  // addr[56:32] | type=2
      i32x8 g1 = {};
      g1[0] = (1 << 16);           // data_size = 2 bytes
      g1[1] = (H_DIM << 16);       // tensor_dim0[15:0] = 64
      g1[2] = (32 << 16);          // tensor_dim1[15:0] = 32
      g1[3] = (H_DIM << 16);       // tile_dim0 = 64
      g1[4] = 32;                  // tile_dim1 = 32
      g1[5] = H_DIM;               // tensor_dim0_stride = 64
      i32x4 z4 = {};
      asm volatile("s_wait_dscnt 0x0" ::: "memory");     // WAR vs prior kbuf reads
#if defined(__clang_major__) && __clang_major__ >= 23
      i32x8 z8 = {};
      __builtin_amdgcn_tensor_load_to_lds(g0, g1, z4, z4, z8, 0);
#else
      __builtin_amdgcn_tensor_load_to_lds(g0, g1, z4, z4, 0);
#endif
      __builtin_amdgcn_s_wait_tensorcnt(0);
      asm volatile("" ::: "memory");
    }
    const _Float16* kp = kbuf;
#else
    const _Float16* kp = km + (size_t)(b * T_DIM + k0) * H_DIM;
#endif

    // K^T B operands: [tile-of-16-keys][K-chunk]; contiguous 32B each
    v16h kb[2][2];
    #pragma unroll
    for (int nt = 0; nt < 2; ++nt)
      #pragma unroll
      for (int c = 0; c < 2; ++c)
        kb[nt][c] = *(const v16h*)(kp + (nt * 16 + col) * H_DIM + c * 32 + khalf);

    v8f s0 = zero, s1 = zero;
    s0 = wmma16(qa[0], kb[0][0], s0);
    s0 = wmma16(qa[1], kb[0][1], s0);
    s1 = wmma16(qa[0], kb[1][0], s1);
    s1 = wmma16(qa[1], kb[1][1], s1);

    const bool diag = (p == npairs - 1);   // only last pair straddles diagonal
    #pragma unroll
    for (int r = 0; r < 8; ++r) {
      const int qrow = r + 8 * hi;
      const int qg   = qt * 16 + qrow;
      float a0 = s0[r] * scale;
      float a1 = s1[r] * scale;
      if (diag) {
        if (k0 + col > qg)      a0 = NEG_INF;
        if (k0 + 16 + col > qg) a1 = NEG_INF;
      }
      float rm = fmaxf(a0, a1);                       // row max over 32 keys
      rm = fmaxf(rm, __shfl_xor(rm, 1, 16));
      rm = fmaxf(rm, __shfl_xor(rm, 2, 16));
      rm = fmaxf(rm, __shfl_xor(rm, 4, 16));
      rm = fmaxf(rm, __shfl_xor(rm, 8, 16));
      float mnew = fmaxf(m_run[r], rm);
      float corr = __expf(m_run[r] - mnew);
      float p0 = __expf(a0 - mnew);
      float p1 = __expf(a1 - mnew);
      float rs = p0 + p1;                             // row sum
      rs += __shfl_xor(rs, 1, 16);
      rs += __shfl_xor(rs, 2, 16);
      rs += __shfl_xor(rs, 4, 16);
      rs += __shfl_xor(rs, 8, 16);
      l_run[r] = l_run[r] * corr + rs;
      m_run[r] = mnew;
      #pragma unroll
      for (int ct = 0; ct < 4; ++ct) o[ct][r] *= corr;
      pbuf[qrow * 32 + col]      = (_Float16)p0;
      pbuf[qrow * 32 + 16 + col] = (_Float16)p1;
    }
    __syncthreads();

    // P (16x32) as full A operand
    v16h pa;
    {
      v8h lo  = *(const v8h*)(pbuf + col * 32 + khalf);
      v8h hi8 = *(const v8h*)(pbuf + col * 32 + khalf + 8);
      #pragma unroll
      for (int j = 0; j < 8; ++j) { pa[j] = lo[j]; pa[8 + j] = hi8[j]; }
    }
    // V^T B operands: contiguous 32B along tokens from transposed V
    #pragma unroll
    for (int ct = 0; ct < 4; ++ct) {
      v16h vb = *(const v16h*)(vt + (size_t)(b * H_DIM + ct * 16 + col) * T_DIM
                                  + k0 + khalf);
      o[ct] = wmma16(pa, vb, o[ct]);
    }
    __syncthreads();
  }

  float* orow = out + (size_t)(b * T_DIM + qt * 16) * H_DIM;
  #pragma unroll
  for (int ct = 0; ct < 4; ++ct)
    #pragma unroll
    for (int r = 0; r < 8; ++r)
      orow[(r + 8 * hi) * H_DIM + ct * 16 + col] = o[ct][r] / l_run[r];
}

extern "C" void kernel_launch(void* const* d_in, const int* in_sizes, int n_in,
                              void* d_out, int out_size, void* d_ws, size_t ws_size,
                              hipStream_t stream) {
  const float* x  = (const float*)d_in[0];
  const float* wq = (const float*)d_in[1];
  const float* wk = (const float*)d_in[2];
  const float* wv = (const float*)d_in[3];
  float* out = (float*)d_out;

  const size_t per = (size_t)B_DIM * T_DIM * H_DIM;     // Q/K/V element count
  _Float16* wt  = (_Float16*)d_ws;                      // 3*64*256 f16 (96 KB)
  _Float16* qb  = wt + (size_t)3 * H_DIM * C_DIM;
  _Float16* kbm = qb + per;
  _Float16* vtm = kbm + per;                            // ~9.5 MB total

  prep_w<<<(3 * H_DIM * C_DIM + 255) / 256, 256, 0, stream>>>(wq, wk, wv, wt);
  qkv_proj<<<(B_DIM * T_DIM) / 16, 32, 0, stream>>>(x, wt, qb, kbm, vtm);
  attn<<<B_DIM * QTILES, 32, 0, stream>>>(qb, kbm, vtm, out);
}
